// HadLinear_61641370632546
// MI455X (gfx1250) — matrix operand order repaired and checked by reading, not compiled, not run on hardware
//
#include <hip/hip_runtime.h>
#include <hip/hip_bf16.h>

typedef unsigned int   u32;
typedef unsigned short u16;
typedef __attribute__((ext_vector_type(16))) __bf16 v16bf;
typedef __attribute__((ext_vector_type(8)))  float  v8f;

#define K_DIM 4096
#define N_DIM 4096
#define BS    1024
#define BM    256
#define BN    128
#define BK    32
#define BKP   40           // padded K stride in ushorts (80B rows, 16B-aligned)
#define NKSTEP (K_DIM / BK)

// ---- bf16 helpers (bit-level, no __bf16 arithmetic) ----
__device__ __forceinline__ u16 f2bf_rne(float f) {
    u32 u = __float_as_uint(f);
    u32 r = u + 0x7FFFu + ((u >> 16) & 1u);
    return (u16)(r >> 16);
}
__device__ __forceinline__ float bf2f(u16 h) {
    return __uint_as_float(((u32)h) << 16);
}

// ---- CDNA5 async global->LDS (ASYNCcnt-tracked, no VGPR staging) ----
__device__ __forceinline__ void async_b128(u32 lds_off, const u16* g) {
    asm volatile("global_load_async_to_lds_b128 %0, %1, off"
                 :: "v"(lds_off), "v"(g) : "memory");
}
__device__ __forceinline__ void wait_async0() {
    asm volatile("s_wait_asynccnt 0" ::: "memory");
}
#define LDS_OFF(p) ((u32)(size_t)(p))   // generic LDS addr: low 32 bits = LDS offset

// ============================================================
// Kernel 1: blockwise FWHT (1024) + scale + bf16 hi/lo split
// ============================================================
__global__ __launch_bounds__(256) void fwht_split_kernel(
        const float* __restrict__ x, u16* __restrict__ xh_hi, u16* __restrict__ xh_lo) {
    __shared__ float s[BS];
    const int tid = threadIdx.x;
    const long long base = (long long)blockIdx.x * BS;

    float4 v = ((const float4*)(x + base))[tid];
    s[tid * 4 + 0] = v.x;
    s[tid * 4 + 1] = v.y;
    s[tid * 4 + 2] = v.z;
    s[tid * 4 + 3] = v.w;

    for (int h = 1; h < BS; h <<= 1) {
        __syncthreads();
        #pragma unroll
        for (int q = 0; q < 2; ++q) {
            int p = tid + q * 256;                       // 512 pairs total
            int i = ((p & ~(h - 1)) << 1) | (p & (h - 1));
            int j = i + h;
            float a = s[i], b = s[j];
            s[i] = a + b;
            s[j] = a - b;
        }
    }
    __syncthreads();

    ushort4 hv, lv;
    {
        float f0 = s[tid * 4 + 0] * 0.03125f;
        float f1 = s[tid * 4 + 1] * 0.03125f;
        float f2 = s[tid * 4 + 2] * 0.03125f;
        float f3 = s[tid * 4 + 3] * 0.03125f;
        hv.x = f2bf_rne(f0); hv.y = f2bf_rne(f1); hv.z = f2bf_rne(f2); hv.w = f2bf_rne(f3);
        lv.x = f2bf_rne(f0 - bf2f(hv.x));
        lv.y = f2bf_rne(f1 - bf2f(hv.y));
        lv.z = f2bf_rne(f2 - bf2f(hv.z));
        lv.w = f2bf_rne(f3 - bf2f(hv.w));
    }
    ((ushort4*)(xh_hi + base))[tid] = hv;
    ((ushort4*)(xh_lo + base))[tid] = lv;
}

// ============================================================
// Kernel 2: split weight f32 -> bf16 hi/lo planes
// ============================================================
__global__ __launch_bounds__(256) void wsplit_kernel(
        const float* __restrict__ w, u16* __restrict__ whi, u16* __restrict__ wlo) {
    const long long i = ((long long)blockIdx.x * 256 + threadIdx.x) * 4;
    float4 v = *(const float4*)(w + i);
    ushort4 hv, lv;
    hv.x = f2bf_rne(v.x); hv.y = f2bf_rne(v.y); hv.z = f2bf_rne(v.z); hv.w = f2bf_rne(v.w);
    lv.x = f2bf_rne(v.x - bf2f(hv.x));
    lv.y = f2bf_rne(v.y - bf2f(hv.y));
    lv.z = f2bf_rne(v.z - bf2f(hv.z));
    lv.w = f2bf_rne(v.w - bf2f(hv.w));
    *(ushort4*)(whi + i) = hv;
    *(ushort4*)(wlo + i) = lv;
}

// ============================================================
// Kernel 3: bf16x3 WMMA GEMM  C[m,n] = sum_k A[m,k]*W[n,k]
//   block tile 256x128x32, 8 waves, 64x64 per wave (4x4 WMMA tiles)
// ============================================================
// A fragment (16-bit A 16x32): lanes 0-15 hold K {0..7, 16..23},
// lanes 16-31 hold K {8..15, 24..31} of row M = lane&15.
__device__ __forceinline__ v16bf ld_fragA(const u16* __restrict__ row, int half) {
    union { uint4 u[2]; v16bf v; } t;
    t.u[0] = *(const uint4*)(row + 8 * half);
    t.u[1] = *(const uint4*)(row + 16 + 8 * half);
    return t.v;
}
// B fragment (32x16): lanes 0-15 hold K {0..7, 16..23}, lanes 16-31 hold K {8..15, 24..31} of column N = lane&15
// (columns of B == rows of W, K-contiguous in memory).
__device__ __forceinline__ v16bf ld_fragB(const u16* __restrict__ row, int half) {
    union { uint4 u[2]; v16bf v; } t;
    t.u[0] = *(const uint4*)(row + 8 * half);
    t.u[1] = *(const uint4*)(row + 16 + 8 * half);
    return t.v;
}

__global__ __launch_bounds__(256) void had_gemm_kernel(
        const u16* __restrict__ Ahi, const u16* __restrict__ Alo,
        const u16* __restrict__ Bhi, const u16* __restrict__ Blo,
        float* __restrict__ C) {
    __shared__ __align__(16) u16 lsA[2][2][BM][BKP];   // [buf][plane][m][k]  80 KiB
    __shared__ __align__(16) u16 lsB[2][2][BN][BKP];   // [buf][plane][n][k]  40 KiB

    const int tid  = threadIdx.x;
    const int lane = tid & 31;
    const int wid  = tid >> 5;
    const int wm   = wid & 3;     // 4 waves along M: 64 rows each
    const int wn   = wid >> 2;    // 2 waves along N: 64 cols each
    const int m0   = blockIdx.y * BM;
    const int n0   = blockIdx.x * BN;

    // loader mapping: A -> thread owns full 32-k row `tid` (0..255), both planes
    //                 B -> thread owns half-row (tid>>1, 16 k), both planes
    const u16* gAh = Ahi + (size_t)(m0 + tid) * K_DIM;
    const u16* gAl = Alo + (size_t)(m0 + tid) * K_DIM;
    const int  brow_ld = tid >> 1;
    const int  bk_ld   = (tid & 1) * 16;
    const u16* gBh = Bhi + (size_t)(n0 + brow_ld) * K_DIM + bk_ld;
    const u16* gBl = Blo + (size_t)(n0 + brow_ld) * K_DIM + bk_ld;

    auto load_tile = [&](int buf, long long ko) {
        const u32 a0 = LDS_OFF(&lsA[buf][0][tid][0]);
        const u32 a1 = LDS_OFF(&lsA[buf][1][tid][0]);
        #pragma unroll
        for (int kk = 0; kk < BK; kk += 8) {
            async_b128(a0 + kk * 2, gAh + ko + kk);
            async_b128(a1 + kk * 2, gAl + ko + kk);
        }
        const u32 b0 = LDS_OFF(&lsB[buf][0][brow_ld][bk_ld]);
        const u32 b1 = LDS_OFF(&lsB[buf][1][brow_ld][bk_ld]);
        #pragma unroll
        for (int kk = 0; kk < 16; kk += 8) {
            async_b128(b0 + kk * 2, gBh + ko + kk);
            async_b128(b1 + kk * 2, gBl + ko + kk);
        }
    };

    v8f acc[4][4];
    #pragma unroll
    for (int mt = 0; mt < 4; ++mt)
        #pragma unroll
        for (int nt = 0; nt < 4; ++nt)
            #pragma unroll
            for (int e = 0; e < 8; ++e)
                acc[mt][nt][e] = 0.0f;

    load_tile(0, 0);
    wait_async0();
    __syncthreads();

    const int arow = wm * 64 + (lane & 15);
    const int brow = wn * 64 + (lane & 15);
    const int half = lane >> 4;

    for (int ks = 0; ks < NKSTEP; ++ks) {
        const int cur = ks & 1;
        if (ks + 1 < NKSTEP)
            load_tile(cur ^ 1, (long long)(ks + 1) * BK);

        v16bf fAh[4], fAl[4], fBh[4], fBl[4];
        #pragma unroll
        for (int mt = 0; mt < 4; ++mt) {
            fAh[mt] = ld_fragA(&lsA[cur][0][arow + mt * 16][0], half);
            fAl[mt] = ld_fragA(&lsA[cur][1][arow + mt * 16][0], half);
        }
        #pragma unroll
        for (int nt = 0; nt < 4; ++nt) {
            fBh[nt] = ld_fragB(&lsB[cur][0][brow + nt * 16][0], half);
            fBl[nt] = ld_fragB(&lsB[cur][1][brow + nt * 16][0], half);
        }

        #pragma unroll
        for (int mt = 0; mt < 4; ++mt) {
            #pragma unroll
            for (int nt = 0; nt < 4; ++nt) {
                acc[mt][nt] = __builtin_amdgcn_wmma_f32_16x16x32_bf16(
                    false, fAh[mt], false, fBh[nt], (short)0, acc[mt][nt], false, false);
                acc[mt][nt] = __builtin_amdgcn_wmma_f32_16x16x32_bf16(
                    false, fAh[mt], false, fBl[nt], (short)0, acc[mt][nt], false, false);
                acc[mt][nt] = __builtin_amdgcn_wmma_f32_16x16x32_bf16(
                    false, fAl[mt], false, fBh[nt], (short)0, acc[mt][nt], false, false);
            }
        }

        wait_async0();
        __syncthreads();
    }

    // C/D layout: VGPR e -> M = e + (lane>=16 ? 8 : 0), N = lane&15
    const int msel = half * 8;
    const int ncol = n0 + wn * 64 + (lane & 15);
    #pragma unroll
    for (int mt = 0; mt < 4; ++mt) {
        const int mbase = m0 + wm * 64 + mt * 16 + msel;
        #pragma unroll
        for (int nt = 0; nt < 4; ++nt) {
            #pragma unroll
            for (int e = 0; e < 8; ++e) {
                C[(long long)(mbase + e) * N_DIM + (ncol + nt * 16)] = acc[mt][nt][e];
            }
        }
    }
}

// ============================================================
// Host side
// ============================================================
extern "C" void kernel_launch(void* const* d_in, const int* in_sizes, int n_in,
                              void* d_out, int out_size, void* d_ws, size_t ws_size,
                              hipStream_t stream) {
    const float* x = (const float*)d_in[0];   // (4, 2048, 4096) f32
    const float* w = (const float*)d_in[1];   // (4096, 4096) f32
    float* out = (float*)d_out;               // (4, 2048, 4096) f32

    const long long xelems = (long long)in_sizes[0];          // 33,554,432
    const long long welems = (long long)N_DIM * K_DIM;        // 16,777,216
    const int M = (int)(xelems / K_DIM);                      // 8192

    // Workspace layout (192 MiB total):
    u16* xh_hi = (u16*)d_ws;
    u16* xh_lo = xh_hi + xelems;
    u16* w_hi  = xh_lo + xelems;
    u16* w_lo  = w_hi + welems;

    // 1) FWHT + split (one 1024-block per WG)
    fwht_split_kernel<<<(int)(xelems / BS), 256, 0, stream>>>(x, xh_hi, xh_lo);

    // 2) Weight split
    wsplit_kernel<<<(int)(welems / (256 * 4)), 256, 0, stream>>>(w, w_hi, w_lo);

    // 3) bf16x3 WMMA GEMM
    dim3 grid(N_DIM / BN, M / BM);
    had_gemm_kernel<<<grid, 256, 0, stream>>>(xh_hi, xh_lo, w_hi, w_lo, out);
}
